// SelfMutualAttention_79113297592787
// MI455X (gfx1250) — matrix operand order repaired and checked by reading, not compiled, not run on hardware
//
#include <hip/hip_runtime.h>

typedef __attribute__((ext_vector_type(16))) _Float16     v16h;
typedef __attribute__((ext_vector_type(8)))  float        v8f;
typedef __attribute__((ext_vector_type(8)))  unsigned int v8u;

#define BATCH 2
#define CCH   256
#define NPIX  4096
#define NHEAD 8
#define DHEAD 32
#define NKT   (NPIX / 32)    // 128 key tiles

// 1 = use gfx1250 async global->LDS copies (ASYNCcnt path); 0 = regular load + ds_store
#define USE_ASYNC_LDS_COPY 1

// A-matrix (16x32 f16) per-lane half layout: vgpr j holds K pair
// {2j, 2j+1} (+16 for lanes 16..31), matching the contiguous B layout
static __device__ __forceinline__ int a_half_base(int j, int grp) {
    int base = 2 * j;
    return base + grp * 16;
}

#if USE_ASYNC_LDS_COPY
static __device__ __forceinline__ void async_copy_b128(void* lds_dst, const void* gsrc) {
    unsigned lds_off = (unsigned)(size_t)lds_dst;               // low 32 bits = LDS offset
    unsigned long long ga = (unsigned long long)(size_t)gsrc;
    asm volatile("global_load_async_to_lds_b128 %0, %1, off"
                 :: "v"(lds_off), "v"(ga) : "memory");
}
static __device__ __forceinline__ void wait_async() {
    asm volatile("s_wait_asynccnt 0x0" ::: "memory");
}
#endif

// ---------------- conversion kernels ----------------

__global__ __launch_bounds__(256) void cvt_f32_to_f16(const float* __restrict__ src,
                                                      _Float16* __restrict__ dst, int n) {
    int i = blockIdx.x * blockDim.x + threadIdx.x;
    if (i < n) dst[i] = (_Float16)src[i];
}

// x [b, c, n] f32  ->  xh [b, n, c] f16  (channel-contiguous rows for B operands)
__global__ __launch_bounds__(256) void transpose_cvt_x(const float* __restrict__ x,
                                                       _Float16* __restrict__ xh) {
    int i = blockIdx.x * blockDim.x + threadIdx.x;
    if (i >= BATCH * CCH * NPIX) return;
    int p = i % NPIX;
    int c = (i / NPIX) % CCH;
    int b = i / (NPIX * CCH);
    xh[((size_t)b * NPIX + p) * CCH + c] = (_Float16)x[i];
}

// ---------------- QKV projection (WMMA GEMM) ----------------
// q,k stored [b,H,n,d]; v stored transposed [b,H,d,n]; 1/sqrt(d) folded into q.

__global__ __launch_bounds__(256) void qkv_proj(
    const _Float16* __restrict__ xh,
    const _Float16* __restrict__ wqh, const float* __restrict__ bq,
    const _Float16* __restrict__ wkh, const float* __restrict__ bk,
    const _Float16* __restrict__ wvh, const float* __restrict__ bv,
    _Float16* __restrict__ qws, _Float16* __restrict__ kws, _Float16* __restrict__ vtws) {

    const int lane = threadIdx.x & 31;
    const int wv   = threadIdx.x >> 5;
    const int grp  = lane >> 4;
    const int l16  = lane & 15;

    int tile = blockIdx.x * 8 + wv;       // 0..8191 = b(2) x Mt(16) x Nt(256)
    int b  = tile >> 12;
    int mt = (tile >> 8) & 15;
    int nt = tile & 255;
    int o0 = mt * 16, p0 = nt * 16;

    const unsigned int* qrow = (const unsigned int*)(wqh + (size_t)(o0 + l16) * CCH);
    const unsigned int* krow = (const unsigned int*)(wkh + (size_t)(o0 + l16) * CCH);
    const unsigned int* vrow = (const unsigned int*)(wvh + (size_t)(o0 + l16) * CCH);
    const unsigned int* xrow = (const unsigned int*)(xh + ((size_t)b * NPIX + p0 + l16) * CCH);

    v8f accq = {}, acck = {}, accv = {};
    for (int k0 = 0; k0 < CCH; k0 += 32) {
        v8u auq, auk, auv, bu;
#pragma unroll
        for (int j = 0; j < 8; ++j) {
            int ah = (k0 + a_half_base(j, grp)) >> 1;
            auq[j] = qrow[ah];
            auk[j] = krow[ah];
            auv[j] = vrow[ah];
            bu[j]  = xrow[(k0 >> 1) + grp * 8 + j];
        }
        v16h aq = __builtin_bit_cast(v16h, auq);
        v16h ak = __builtin_bit_cast(v16h, auk);
        v16h av = __builtin_bit_cast(v16h, auv);
        v16h bx = __builtin_bit_cast(v16h, bu);
        accq = __builtin_amdgcn_wmma_f32_16x16x32_f16(false, aq, false, bx, (short)0, accq, false, false);
        acck = __builtin_amdgcn_wmma_f32_16x16x32_f16(false, ak, false, bx, (short)0, acck, false, false);
        accv = __builtin_amdgcn_wmma_f32_16x16x32_f16(false, av, false, bx, (short)0, accv, false, false);
    }

    const float qscale = 0.17677669529663687f;   // 1/sqrt(32)
#pragma unroll
    for (int r = 0; r < 8; ++r) {
        int o    = o0 + r + grp * 8;
        int head = o >> 5, dim = o & 31;
        int pix  = p0 + l16;
        float qv = (accq[r] + bq[o]) * qscale;
        float kv =  acck[r] + bk[o];
        float vv =  accv[r] + bv[o];
        size_t base = ((size_t)(b * NHEAD + head) * NPIX + pix) * DHEAD + dim;
        qws[base] = (_Float16)qv;
        kws[base] = (_Float16)kv;
        vtws[((size_t)(b * NHEAD + head) * DHEAD + dim) * NPIX + pix] = (_Float16)vv;
    }
}

// ---------------- Flash attention ----------------
// One block = 8 waves = 128 consecutive query rows of one batch-head.
// K/V tiles (32 keys) double-buffered in LDS via async global->LDS copies:
// tile i+1 streams in (ASYNCcnt) while tile i is consumed by WMMAs; one
// barrier per iteration both publishes the next buffer and retires the old.

__global__ __launch_bounds__(256) void flash_attn(
    const _Float16* __restrict__ qws, const _Float16* __restrict__ kws,
    const _Float16* __restrict__ vtws, _Float16* __restrict__ attnh) {

    __shared__ __align__(16) _Float16 kbuf[2][32][32];   // [buf][key][dim] 4 KB
    __shared__ __align__(16) _Float16 vbuf[2][32][32];   // [buf][dim][key] 4 KB
    __shared__ __align__(16) _Float16 plds[8][16][32];   // per-wave P staging 8 KB

    const int tid  = threadIdx.x;
    const int lane = tid & 31;
    const int wv   = tid >> 5;
    const int grp  = lane >> 4;
    const int l16  = lane & 15;

    const int bh = blockIdx.x >> 5;                       // 512 blocks, 32 per bh
    const int q0 = ((blockIdx.x & 31) * 8 + wv) * 16;
    const int b  = bh >> 3, head = bh & 7;

    const _Float16* kbase = kws  + (size_t)bh * NPIX * DHEAD;
    const _Float16* vbase = vtws + (size_t)bh * DHEAD * NPIX;

    // cooperative-copy assignments (fixed per thread): threads 0..127 -> K tile,
    // threads 128..255 -> V tile; 16 bytes each => full 4 KB per round.
    const int is_v  = tid >> 7;                           // 0: K, 1: V
    const int tloc  = tid & 127;
    const int vdim  = tloc >> 2, vpart = tloc & 3;

    // Q tile as A operand (reused every iteration)
    const unsigned int* qrow = (const unsigned int*)(qws + ((size_t)bh * NPIX + q0 + l16) * DHEAD);
    v8u qu;
#pragma unroll
    for (int j = 0; j < 8; ++j) qu[j] = qrow[a_half_base(j, grp) >> 1];
    v16h qa = __builtin_bit_cast(v16h, qu);

    // all-ones B matrix: lacc = P x ones + lacc  -> per-lane running row sums
    v16h ones;
#pragma unroll
    for (int i = 0; i < 16; ++i) ones[i] = (_Float16)1.0f;

    v8f oacc0 = {}, oacc1 = {}, lacc = {};
    float mrun[8];
#pragma unroll
    for (int r = 0; r < 8; ++r) mrun[r] = -3.0e38f;

    // stage tile 0 into buffer 0
    {
        const _Float16* gsrc = is_v ? vbase + (size_t)vdim * NPIX + vpart * 8
                                    : kbase + tloc * 8;
        _Float16* ldst = is_v ? &vbuf[0][0][0] + tloc * 8 : &kbuf[0][0][0] + tloc * 8;
#if USE_ASYNC_LDS_COPY
        async_copy_b128(ldst, gsrc);
        wait_async();
#else
        *(uint4*)ldst = *(const uint4*)gsrc;
#endif
    }
    __syncthreads();

    for (int it = 0; it < NKT; ++it) {
        const int cur = it & 1;
        const int kt  = it * 32;

        // ---- prefetch tile it+1 into the alternate buffer (async, no wait) ----
        if (it + 1 < NKT) {
            int ktn = kt + 32;
            const _Float16* gsrc = is_v ? vbase + (size_t)vdim * NPIX + ktn + vpart * 8
                                        : kbase + (size_t)ktn * DHEAD + tloc * 8;
            _Float16* ldst = is_v ? &vbuf[cur ^ 1][0][0] + tloc * 8
                                  : &kbuf[cur ^ 1][0][0] + tloc * 8;
#if USE_ASYNC_LDS_COPY
            async_copy_b128(ldst, gsrc);
#else
            *(uint4*)ldst = *(const uint4*)gsrc;
#endif
        }

        // ---- S = Q K^T  (16 x 32 via two WMMAs, B from LDS) ----
        const unsigned int* kr0 = (const unsigned int*)&kbuf[cur][l16][0];
        const unsigned int* kr1 = (const unsigned int*)&kbuf[cur][16 + l16][0];
        v8u b0u, b1u;
#pragma unroll
        for (int j = 0; j < 8; ++j) { b0u[j] = kr0[grp * 8 + j]; b1u[j] = kr1[grp * 8 + j]; }
        v16h kb0 = __builtin_bit_cast(v16h, b0u);
        v16h kb1 = __builtin_bit_cast(v16h, b1u);
        v8f s0 = {}, s1 = {};
        s0 = __builtin_amdgcn_wmma_f32_16x16x32_f16(false, qa, false, kb0, (short)0, s0, false, false);
        s1 = __builtin_amdgcn_wmma_f32_16x16x32_f16(false, qa, false, kb1, (short)0, s1, false, false);

        // ---- online softmax: only the row-max needs cross-lane reduction ----
#pragma unroll
        for (int r = 0; r < 8; ++r) {
            float rm = fmaxf(s0[r], s1[r]);
            rm = fmaxf(rm, __shfl_xor(rm, 1, 32));
            rm = fmaxf(rm, __shfl_xor(rm, 2, 32));
            rm = fmaxf(rm, __shfl_xor(rm, 4, 32));
            rm = fmaxf(rm, __shfl_xor(rm, 8, 32));
            float mo = mrun[r];
            float mn = fmaxf(mo, rm);
            float sc = __expf(mo - mn);
            float p0 = __expf(s0[r] - mn);
            float p1 = __expf(s1[r] - mn);
            mrun[r] = mn;
            oacc0[r] *= sc;
            oacc1[r] *= sc;
            lacc[r]  *= sc;
            plds[wv][r + grp * 8][l16]      = (_Float16)p0;
            plds[wv][r + grp * 8][16 + l16] = (_Float16)p1;
        }

        // ---- P as A operand (same-wave LDS ops are in-order) ----
        const unsigned int* prow = (const unsigned int*)&plds[wv][l16][0];
        v8u pu;
#pragma unroll
        for (int j = 0; j < 8; ++j) pu[j] = prow[a_half_base(j, grp) >> 1];
        v16h pa = __builtin_bit_cast(v16h, pu);

        // ---- O += P*V (dims 0..15 / 16..31), l += P*ones ----
        const unsigned int* vr0 = (const unsigned int*)&vbuf[cur][l16][0];
        const unsigned int* vr1 = (const unsigned int*)&vbuf[cur][16 + l16][0];
        v8u v0u, v1u;
#pragma unroll
        for (int j = 0; j < 8; ++j) { v0u[j] = vr0[grp * 8 + j]; v1u[j] = vr1[grp * 8 + j]; }
        v16h vb0 = __builtin_bit_cast(v16h, v0u);
        v16h vb1 = __builtin_bit_cast(v16h, v1u);
        oacc0 = __builtin_amdgcn_wmma_f32_16x16x32_f16(false, pa, false, vb0, (short)0, oacc0, false, false);
        oacc1 = __builtin_amdgcn_wmma_f32_16x16x32_f16(false, pa, false, vb1, (short)0, oacc1, false, false);
        lacc  = __builtin_amdgcn_wmma_f32_16x16x32_f16(false, pa, false, ones, (short)0, lacc,  false, false);

        // ---- publish next buffer / retire current (single barrier) ----
        if (it + 1 < NKT) {
#if USE_ASYNC_LDS_COPY
            wait_async();
#endif
            __syncthreads();
        }
    }

    // ---- finalize: attnh [b, n, c] f16 ----
#pragma unroll
    for (int r = 0; r < 8; ++r) {
        int row = r + grp * 8;
        float inv = 1.0f / lacc[r];
        size_t obase = ((size_t)b * NPIX + q0 + row) * CCH + head * DHEAD;
        attnh[obase + l16]      = (_Float16)(oacc0[r] * inv);
        attnh[obase + 16 + l16] = (_Float16)(oacc1[r] * inv);
    }
}

// ---------------- output projection ----------------

__global__ __launch_bounds__(256) void out_proj(
    const _Float16* __restrict__ attnh, const _Float16* __restrict__ woh,
    const float* __restrict__ bo, float* __restrict__ out) {

    const int lane = threadIdx.x & 31;
    const int wv   = threadIdx.x >> 5;
    const int grp  = lane >> 4;
    const int l16  = lane & 15;

    int tile = blockIdx.x * 8 + wv;       // b(2) x Mt(16) x Nt(256)
    int b  = tile >> 12;
    int mt = (tile >> 8) & 15;
    int nt = tile & 255;
    int o0 = mt * 16, p0 = nt * 16;

    const unsigned int* wrow = (const unsigned int*)(woh + (size_t)(o0 + l16) * CCH);
    const unsigned int* arow = (const unsigned int*)(attnh + ((size_t)b * NPIX + p0 + l16) * CCH);

    v8f acc = {};
    for (int k0 = 0; k0 < CCH; k0 += 32) {
        v8u au, bu;
#pragma unroll
        for (int j = 0; j < 8; ++j) {
            au[j] = wrow[(k0 + a_half_base(j, grp)) >> 1];
            bu[j] = arow[(k0 >> 1) + grp * 8 + j];
        }
        v16h a  = __builtin_bit_cast(v16h, au);
        v16h bb = __builtin_bit_cast(v16h, bu);
        acc = __builtin_amdgcn_wmma_f32_16x16x32_f16(false, a, false, bb, (short)0, acc, false, false);
    }
#pragma unroll
    for (int r = 0; r < 8; ++r) {
        int o = o0 + r + grp * 8;
        out[((size_t)(b * CCH + o)) * NPIX + p0 + l16] = acc[r] + bo[o];
    }
}

// ---------------- host ----------------

extern "C" void kernel_launch(void* const* d_in, const int* in_sizes, int n_in,
                              void* d_out, int out_size, void* d_ws, size_t ws_size,
                              hipStream_t stream) {
    (void)in_sizes; (void)n_in; (void)out_size; (void)ws_size;
    const float* x1 = (const float*)d_in[0];
    const float* Wq = (const float*)d_in[1];
    const float* bq = (const float*)d_in[2];
    const float* Wk = (const float*)d_in[3];
    const float* bk = (const float*)d_in[4];
    const float* Wv = (const float*)d_in[5];
    const float* bv = (const float*)d_in[6];
    const float* Wo = (const float*)d_in[7];
    const float* bo = (const float*)d_in[8];

    _Float16* ws = (_Float16*)d_ws;
    size_t off = 0;
    _Float16* xh   = ws + off; off += (size_t)BATCH * NPIX * CCH;
    _Float16* wqh  = ws + off; off += (size_t)CCH * CCH;
    _Float16* wkh  = ws + off; off += (size_t)CCH * CCH;
    _Float16* wvh  = ws + off; off += (size_t)CCH * CCH;
    _Float16* woh  = ws + off; off += (size_t)CCH * CCH;
    _Float16* qws  = ws + off; off += (size_t)BATCH * NHEAD * NPIX * DHEAD;
    _Float16* kws  = ws + off; off += (size_t)BATCH * NHEAD * NPIX * DHEAD;
    _Float16* vtws = ws + off; off += (size_t)BATCH * NHEAD * NPIX * DHEAD;
    _Float16* atth = ws + off; off += (size_t)BATCH * NPIX * CCH;

    int nx = BATCH * CCH * NPIX;
    transpose_cvt_x<<<(nx + 255) / 256, 256, 0, stream>>>(x1, xh);
    int nw = CCH * CCH;
    cvt_f32_to_f16<<<(nw + 255) / 256, 256, 0, stream>>>(Wq, wqh, nw);
    cvt_f32_to_f16<<<(nw + 255) / 256, 256, 0, stream>>>(Wk, wkh, nw);
    cvt_f32_to_f16<<<(nw + 255) / 256, 256, 0, stream>>>(Wv, wvh, nw);
    cvt_f32_to_f16<<<(nw + 255) / 256, 256, 0, stream>>>(Wo, woh, nw);

    qkv_proj<<<1024, 256, 0, stream>>>(xh, wqh, bq, wkh, bk, wvh, bv, qws, kws, vtws);
    flash_attn<<<512, 256, 0, stream>>>(qws, kws, vtws, atth);
    out_proj<<<1024, 256, 0, stream>>>(atth, woh, bo, (float*)d_out);
}
